// ChebyKANLayer_4286377362152
// MI455X (gfx1250) — matrix operand order repaired and checked by reading, not compiled, not run on hardware
//
#include <hip/hip_runtime.h>

// ChebyKAN layer fused as a single f16-WMMA GEMM with K = 1024*8 (Chebyshev
// basis x coeffs) + 1024 (x @ W^T), accumulated in f32.
//
//   out[16384,1024] = [T(tanh x) | x]  @  [C_flat | W]^T
//
// Compute-bound (AI ~1900 FLOP/byte), so everything rides the
// v_wmma_f32_16x16x32_f16 pipe; A-matrix (basis) is generated on the fly in
// registers and staged through LDS, B converted f32->f16 while staging.

typedef __attribute__((ext_vector_type(16))) _Float16 v16h;
typedef __attribute__((ext_vector_type(8)))  _Float16 v8h;
typedef __attribute__((ext_vector_type(4)))  _Float16 v4h;
typedef __attribute__((ext_vector_type(8)))  float    v8f;

#define IN_F   1024
#define OUT_F  1024
#define BATCH  16384
#define BM     128   // C macro-tile rows per block
#define BN     128   // C macro-tile cols per block
#define BK     64    // K-tile (KAN: 8 features x 8 degrees)

union V16U { v16h v; v8h h[2]; };

__global__ __launch_bounds__(256) void cheby_kan_fused(
    const float* __restrict__ x,       // [16384, 1024]
    const float* __restrict__ coeffs,  // [1024, 1024, 8]
    const float* __restrict__ bw,      // [1024, 1024]
    float* __restrict__ out)           // [16384, 1024]
{
  __shared__ __attribute__((aligned(16))) _Float16 As[BM * BK]; // 16 KB, row-major [row][k]
  __shared__ __attribute__((aligned(16))) _Float16 Bs[BN * BK]; // 16 KB, [col][k] (K contiguous)

  const int tid  = threadIdx.x;
  const int wave = tid >> 5;          // 8 waves
  const int lane = tid & 31;
  const int l16  = lane & 15;
  const int lh   = lane >> 4;         // lane-half selector (ISA fragment layouts)
  const int m_w  = wave & 3;          // 4 waves along M: 32 rows each
  const int n_w  = wave >> 2;         // 2 waves along N: 64 cols each
  const int blockN = blockIdx.x * BN;
  const int blockM = blockIdx.y * BM;

  v8f acc[2][4] = {};                 // 32x64 per wave = 2x4 16x16 tiles

  const int KAN_TILES  = IN_F / 8;    // 128 tiles of K=64 (8 feats * 8 degrees)
  const int BASE_TILES = IN_F / BK;   // 16 tiles of K=64 over raw x
  const int TOT = KAN_TILES + BASE_TILES;

  for (int t = 0; t < TOT; ++t) {
    __syncthreads();                  // previous tile fully consumed
    const bool kan = (t < KAN_TILES);

    // ---- stage A tile into LDS ----
    if (kan) {
      const int i0 = t * 8;           // first input feature of this K-tile
      #pragma unroll
      for (int q = 0; q < 4; ++q) {   // 128 rows * 8 feats = 1024 (row,i) pairs
        int pair = tid + 256 * q;
        int row  = pair >> 3;
        int il   = pair & 7;
        float xv = x[(size_t)(blockM + row) * IN_F + i0 + il];
        float tn = tanhf(xv);
        v8h b;
        b[0] = (_Float16)1.0f;
        b[1] = (_Float16)tn;
        float tp = 1.0f, tc = tn;
        #pragma unroll
        for (int d = 2; d < 8; ++d) {
          float tn2 = 2.0f * tn * tc - tp;  // Chebyshev recurrence
          tp = tc; tc = tn2;
          b[d] = (_Float16)tn2;
        }
        *(v8h*)&As[row * BK + il * 8] = b;  // ds_store_b128
      }
    } else {
      const int j0 = (t - KAN_TILES) * BK;
      #pragma unroll
      for (int q = 0; q < 8; ++q) {   // 128*64 floats = 2048 float4
        int v   = tid + 256 * q;
        int row = v >> 4;
        int c4  = v & 15;
        float4 f = *(const float4*)&x[(size_t)(blockM + row) * IN_F + j0 + c4 * 4];
        v4h h;
        h[0] = (_Float16)f.x; h[1] = (_Float16)f.y;
        h[2] = (_Float16)f.z; h[3] = (_Float16)f.w;
        *(v4h*)&As[row * BK + c4 * 4] = h;
      }
    }

    // ---- stage B tile into LDS (f32 -> f16) ----
    {
      const float* wsrc;
      int wstride, k0;
      if (kan) { wsrc = coeffs; wstride = IN_F * 8; k0 = t * 64; }          // C[o, i0:i0+8, :] contiguous
      else     { wsrc = bw;     wstride = IN_F;     k0 = (t - KAN_TILES) * BK; }
      #pragma unroll
      for (int q = 0; q < 8; ++q) {   // 128 cols * 64 K = 2048 float4
        int v   = tid + 256 * q;
        int col = v >> 4;
        int c4  = v & 15;
        const float* gp = &wsrc[(size_t)(blockN + col) * wstride + k0 + c4 * 4];
        float4 f = *(const float4*)gp;
        __builtin_prefetch(gp + BK, 0, 1);   // next K-chunk of this column
        v4h h;
        h[0] = (_Float16)f.x; h[1] = (_Float16)f.y;
        h[2] = (_Float16)f.z; h[3] = (_Float16)f.w;
        *(v4h*)&Bs[col * BK + c4 * 4] = h;
      }
    }

    __syncthreads();                  // tiles visible to all waves

    // ---- WMMA over the 64-wide K-tile (two 16x16x32 steps) ----
    #pragma unroll
    for (int kk = 0; kk < BK; kk += 32) {
      v16h af[2], bf[4];
      #pragma unroll
      for (int ms = 0; ms < 2; ++ms) {
        // A 16x32 f16 layout: lane l<16 -> K 0..7 / 16..23, l>=16 -> K 8..15 / 24..31
        const _Float16* p = &As[(m_w * 32 + ms * 16 + l16) * BK + kk + lh * 8];
        V16U u;
        u.h[0] = *(const v8h*)p;          // K +0..7
        u.h[1] = *(const v8h*)(p + 16);   // K +16..23
        af[ms] = u.v;
      }
      #pragma unroll
      for (int ns = 0; ns < 4; ++ns) {
        // B 32x16 f16 layout: lane l<16 -> K 0..7 / 16..23, l>=16 -> K 8..15 / 24..31 (col = l&15)
        const _Float16* p = &Bs[(n_w * 64 + ns * 16 + l16) * BK + kk + lh * 8];
        V16U u;
        u.h[0] = *(const v8h*)p;
        u.h[1] = *(const v8h*)(p + 16);
        bf[ns] = u.v;
      }
      #pragma unroll
      for (int ms = 0; ms < 2; ++ms)
        #pragma unroll
        for (int ns = 0; ns < 4; ++ns)
          acc[ms][ns] = __builtin_amdgcn_wmma_f32_16x16x32_f16(
              false, af[ms], false, bf[ns],
              (short)0, acc[ms][ns], false, false);
    }
  }

  // ---- epilogue: C/D layout — lane l: N = l&15, VGPR r: M = (l>>4)*8 + r ----
  #pragma unroll
  for (int ms = 0; ms < 2; ++ms) {
    #pragma unroll
    for (int ns = 0; ns < 4; ++ns) {
      int col  = blockN + n_w * 64 + ns * 16 + l16;
      int rowb = blockM + m_w * 32 + ms * 16 + lh * 8;
      float* po = out + (size_t)rowb * OUT_F + col;
      #pragma unroll
      for (int r = 0; r < 8; ++r)
        po[(size_t)r * OUT_F] = acc[ms][ns][r];
    }
  }
}

extern "C" void kernel_launch(void* const* d_in, const int* in_sizes, int n_in,
                              void* d_out, int out_size, void* d_ws, size_t ws_size,
                              hipStream_t stream) {
  const float* x      = (const float*)d_in[0];   // [16384,1024]
  const float* coeffs = (const float*)d_in[1];   // [1024,1024,8]
  const float* bw     = (const float*)d_in[2];   // [1024,1024]
  float* out = (float*)d_out;

  dim3 grid(OUT_F / BN, BATCH / BM);  // (8, 128) blocks
  cheby_kan_fused<<<grid, dim3(256), 0, stream>>>(x, coeffs, bw, out);
}
